// coRNN_58712202936373
// MI455X (gfx1250) — compile-verified, matrix operand order checked by reading, not run on hardware
//
#include <hip/hip_runtime.h>
#include <hip/hip_bf16.h>

// ---------------------------------------------------------------------------
// coRNN on gfx1250:
//  1) prep: f32->f16 weights + fused bias into d_ws
//  2) proj: Vx = x @ V_w^T + V_b via v_wmma_f32_16x16x32_f16, written in-place
//     into the hys region of d_out
//  3) recur: batch-split persistent recurrence; LDS-resident state; per-step
//     vx staged ahead of time with global_load_async_to_lds_b128 (ASYNCcnt)
//     double-buffered in LDS so no global-load latency sits on the serial path
// ---------------------------------------------------------------------------

typedef __attribute__((ext_vector_type(16))) _Float16 v16h;
typedef __attribute__((ext_vector_type(8)))  _Float16 h8;
typedef __attribute__((ext_vector_type(8)))  float    v8f;
typedef __attribute__((ext_vector_type(4)))  float    f4;

#define BB   64
#define TT   1024
#define II   128
#define HH   512
#define BSUB 16      // batches per workgroup (== WMMA M)
#define DT_C    0.042f
#define GAMMA_C 2.7f
#define EPS_C   4.7f

static __device__ __forceinline__ v8f wmma16(v16h a, v16h b, v8f c) {
    return __builtin_amdgcn_wmma_f32_16x16x32_f16(
        /*neg_a=*/false, a, /*neg_b=*/false, b,
        /*c_mod=*/(short)0, c, /*reuse_a=*/false, /*reuse_b=*/false);
}

static __device__ __forceinline__ v16h cat8(h8 lo, h8 hi) {
    return __builtin_shufflevector(lo, hi,
        0,1,2,3,4,5,6,7,8,9,10,11,12,13,14,15);
}

// Async copy 16 bytes: global -> LDS, tracked by ASYNCcnt.
static __device__ __forceinline__ void async_g2l_b128(unsigned lds_off,
                                                      const float* gaddr) {
    asm volatile("global_load_async_to_lds_b128 %0, %1, off"
                 :
                 : "v"(lds_off), "v"((unsigned long long)(uintptr_t)gaddr)
                 : "memory");
}

static __device__ __forceinline__ void wait_async0() {
    asm volatile("s_wait_asynccnt 0x0" ::: "memory");
}

// --- Kernel 1: f32 -> f16 weight conversion + bias fuse ---------------------
__global__ __launch_bounds__(256)
void prep_kernel(const float* __restrict__ wy, const float* __restrict__ wz,
                 const float* __restrict__ vw,
                 const float* __restrict__ wyb, const float* __restrict__ wzb,
                 _Float16* __restrict__ wyh, _Float16* __restrict__ wzh,
                 _Float16* __restrict__ vwh, float* __restrict__ bias)
{
    int i = blockIdx.x * blockDim.x + threadIdx.x;
    if (i < HH * HH) {
        wyh[i] = (_Float16)wy[i];
        wzh[i] = (_Float16)wz[i];
    }
    if (i < HH * II) vwh[i] = (_Float16)vw[i];
    if (i < HH)      bias[i] = wyb[i] + wzb[i];
}

// --- Kernel 2: Vx = x @ V_w^T + V_b, written into hys region of d_out -------
// GEMM: M = B*T = 65536, K = 128, N = 512. One wave per 16x16 tile.
__global__ __launch_bounds__(256)
void proj_kernel(const float* __restrict__ x, const _Float16* __restrict__ vwh,
                 const float* __restrict__ vb, float* __restrict__ out)
{
    const int lane = threadIdx.x & 31;
    const int wave = threadIdx.x >> 5;
    const int tile = blockIdx.x * 8 + wave;          // 131072 tiles total
    const int mt = tile >> 5;                        // 4096 M-tiles
    const int nt = tile & 31;                        // 32 N-tiles
    const int m0 = mt * 16, n0 = nt * 16;
    const int hi = lane >> 4, l16 = lane & 15;
    const int m = m0 + l16;                          // A row for this lane
    const int n = n0 + l16;                          // B/D column for this lane

    v8f acc = {};
#pragma unroll
    for (int kc = 0; kc < II; kc += 32) {
        // A operand: x[m][k] f32 -> f16, two contiguous 8-float runs per lane
        const float* pa = x + (size_t)m * II + kc + 8 * hi;
        f4 a00 = *(const f4*)(pa + 0);
        f4 a01 = *(const f4*)(pa + 4);
        f4 a10 = *(const f4*)(pa + 16);
        f4 a11 = *(const f4*)(pa + 20);
        v16h A;
#pragma unroll
        for (int e = 0; e < 4; ++e) {
            A[e]      = (_Float16)a00[e];
            A[e + 4]  = (_Float16)a01[e];
            A[e + 8]  = (_Float16)a10[e];
            A[e + 12] = (_Float16)a11[e];
        }
        // B operand: V_w f16 row-major [h][i]; two 16B vector loads per lane
        const _Float16* pb = vwh + (size_t)n * II + kc + 8 * hi;
        v16h Bm = cat8(*(const h8*)pb, *(const h8*)(pb + 16));
        acc = wmma16(A, Bm, acc);
    }

    const float b = vb[n];
#pragma unroll
    for (int r = 0; r < 8; ++r) {
        const int mm = m0 + r + 8 * hi;              // D: m = r + 8*hi
        out[(size_t)mm * HH + n] = acc[r] + b;       // hys[(b*T+t)][h] slot
    }
}

// --- Kernel 3: sequential recurrence, 4 WGs x 16 batches, 16 waves each -----
__global__ __launch_bounds__(512)
void recur_kernel(const float* __restrict__ hy0, const float* __restrict__ hz0,
                  const _Float16* __restrict__ wyh, const _Float16* __restrict__ wzh,
                  const float* __restrict__ bias, float* __restrict__ out)
{
    __shared__ float    hyf[BSUB][HH];     // f32 master state      (32 KB)
    __shared__ float    hzf[BSUB][HH];     //                       (32 KB)
    __shared__ _Float16 hyh[BSUB][HH];     // f16 WMMA A-operands   (16 KB)
    __shared__ _Float16 hzh[BSUB][HH];     //                       (16 KB)
    __shared__ float    vxbuf[2][BSUB][HH];// vx double buffer      (64 KB)

    const int tid   = threadIdx.x;
    const int bbase = blockIdx.x * BSUB;

    for (int i = tid; i < BSUB * HH; i += 512) {
        const int bb = i >> 9, k = i & (HH - 1);
        const float y = hy0[(size_t)(bbase + bb) * HH + k];
        const float z = hz0[(size_t)(bbase + bb) * HH + k];
        hyf[bb][k] = y; hzf[bb][k] = z;
        hyh[bb][k] = (_Float16)y; hzh[bb][k] = (_Float16)z;
    }

    // Stage vx for t=0 into vxbuf[0] (async, 4 x b128 per thread).
    // Load index li covers floats [4*li, 4*li+4): bb = li>>7, k = (li&127)*4.
    const unsigned lds0 = (unsigned)(uintptr_t)&vxbuf[0][0][0];
    const unsigned lds1 = (unsigned)(uintptr_t)&vxbuf[1][0][0];
#pragma unroll
    for (int j = 0; j < 4; ++j) {
        const int li = tid + j * 512;
        const int bb = li >> 7, k = (li & 127) << 2;
        const float* g = out + ((size_t)(bbase + bb) * TT + 0) * HH + k;
        async_g2l_b128(lds0 + (unsigned)li * 16u, g);
    }
    wait_async0();
    __syncthreads();

    const int lane = tid & 31, wave = tid >> 5;      // 16 waves, 2 N-tiles each
    const int hi = lane >> 4, l16 = lane & 15;

    for (int t = 0; t < TT; ++t) {
        const int cur = t & 1;
        const unsigned ldsn = cur ? lds0 : lds1;     // prefetch target = other buf

        // --- prefetch vx(t+1) into the other buffer; overlaps the matmuls ---
        if (t + 1 < TT) {
#pragma unroll
            for (int j = 0; j < 4; ++j) {
                const int li = tid + j * 512;
                const int bb = li >> 7, k = (li & 127) << 2;
                const float* g = out + ((size_t)(bbase + bb) * TT + (t + 1)) * HH + k;
                async_g2l_b128(ldsn + (unsigned)li * 16u, g);
            }
        }

        // --- matmul phase: pre-acts for this wave's two 16-wide h tiles -----
        v8f acc0 = {}, acc1 = {};
        const int n0 = (wave * 2 + 0) * 16 + l16;
        const int n1 = (wave * 2 + 1) * 16 + l16;

        for (int kc = 0; kc < HH; kc += 32) {
            const _Float16* pay = &hyh[l16][kc + 8 * hi];
            const _Float16* paz = &hzh[l16][kc + 8 * hi];
            v16h Ay = cat8(*(const h8*)pay, *(const h8*)(pay + 16));
            v16h Az = cat8(*(const h8*)paz, *(const h8*)(paz + 16));

            const _Float16* pby0 = wyh + (size_t)n0 * HH + kc + 8 * hi;
            const _Float16* pbz0 = wzh + (size_t)n0 * HH + kc + 8 * hi;
            acc0 = wmma16(Ay, cat8(*(const h8*)pby0, *(const h8*)(pby0 + 16)), acc0);
            acc0 = wmma16(Az, cat8(*(const h8*)pbz0, *(const h8*)(pbz0 + 16)), acc0);

            const _Float16* pby1 = wyh + (size_t)n1 * HH + kc + 8 * hi;
            const _Float16* pbz1 = wzh + (size_t)n1 * HH + kc + 8 * hi;
            acc1 = wmma16(Ay, cat8(*(const h8*)pby1, *(const h8*)(pby1 + 16)), acc1);
            acc1 = wmma16(Az, cat8(*(const h8*)pbz1, *(const h8*)(pbz1 + 16)), acc1);
        }
        __syncthreads();   // all waves done reading LDS state for step t

        // --- update phase: vx from LDS (pre-staged), Euler step, store hy ---
        const float bs0 = bias[n0];
        const float bs1 = bias[n1];
#pragma unroll
        for (int r = 0; r < 8; ++r) {
            const int bb = r + 8 * hi;               // local batch 0..15
            const size_t row = ((size_t)(bbase + bb) * TT + t) * HH;
            {
                const float vx  = vxbuf[cur][bb][n0];
                const float pre = tanhf(acc0[r] + bs0 + vx);
                const float y = hyf[bb][n0], z = hzf[bb][n0];
                const float zn = z + DT_C * (pre - GAMMA_C * y - EPS_C * z);
                const float yn = y + DT_C * zn;
                hyf[bb][n0] = yn; hzf[bb][n0] = zn;
                hyh[bb][n0] = (_Float16)yn; hzh[bb][n0] = (_Float16)zn;
                out[row + n0] = yn;                  // hys[b,t,n0]
            }
            {
                const float vx  = vxbuf[cur][bb][n1];
                const float pre = tanhf(acc1[r] + bs1 + vx);
                const float y = hyf[bb][n1], z = hzf[bb][n1];
                const float zn = z + DT_C * (pre - GAMMA_C * y - EPS_C * z);
                const float yn = y + DT_C * zn;
                hyf[bb][n1] = yn; hzf[bb][n1] = zn;
                hyh[bb][n1] = (_Float16)yn; hzh[bb][n1] = (_Float16)zn;
                out[row + n1] = yn;
            }
        }
        wait_async0();     // vx(t+1) staged (latency hidden behind matmuls)
        __syncthreads();   // state + staged vx visible to all waves
    }

    // final hy, hz tails of d_out
    const size_t tail = (size_t)BB * TT * HH;
    for (int i = tid; i < BSUB * HH; i += 512) {
        const int bb = i >> 9, k = i & (HH - 1);
        out[tail + (size_t)(bbase + bb) * HH + k]                    = hyf[bb][k];
        out[tail + (size_t)BB * HH + (size_t)(bbase + bb) * HH + k]  = hzf[bb][k];
    }
}

// ---------------------------------------------------------------------------
extern "C" void kernel_launch(void* const* d_in, const int* in_sizes, int n_in,
                              void* d_out, int out_size, void* d_ws, size_t ws_size,
                              hipStream_t stream) {
    const float* x    = (const float*)d_in[0];   // [B,T,I]
    const float* hy0  = (const float*)d_in[1];   // [B,H]
    const float* hz0  = (const float*)d_in[2];   // [B,H]
    const float* wy_w = (const float*)d_in[3];   // [H,H]
    const float* wy_b = (const float*)d_in[4];   // [H]
    const float* wz_w = (const float*)d_in[5];   // [H,H]
    const float* wz_b = (const float*)d_in[6];   // [H]
    const float* V_w  = (const float*)d_in[7];   // [H,I]
    const float* V_b  = (const float*)d_in[8];   // [H]
    float* out = (float*)d_out;

    _Float16* wyh = (_Float16*)d_ws;
    _Float16* wzh = wyh + (size_t)HH * HH;
    _Float16* vwh = wzh + (size_t)HH * HH;
    float*    bias = (float*)(vwh + (size_t)HH * II);

    // 1) weight conversion / bias fuse
    prep_kernel<<<(HH * HH + 255) / 256, 256, 0, stream>>>(
        wy_w, wz_w, V_w, wy_b, wz_b, wyh, wzh, vwh, bias);

    // 2) input projection: 4096 M-tiles * 32 N-tiles, 8 waves per block
    const int tiles = (BB * TT / 16) * (HH / 16);
    proj_kernel<<<tiles / 8, 256, 0, stream>>>(x, vwh, V_b, out);

    // 3) recurrence: one WG per 16-batch slice
    recur_kernel<<<BB / BSUB, 512, 0, stream>>>(hy0, hz0, wyh, wzh, bias, out);
}
